// RadarRnn1_24558622999000
// MI455X (gfx1250) — compile-verified, matrix operand order checked by reading, not run on hardware
//
#include <hip/hip_runtime.h>
#include <cstdint>
#include <cstddef>

// ---------------------------------------------------------------------------
// RadarRnn1: 4-layer tanh-RNN (B=512, S=1024, IN=64, H=32) + per-step Linear
// + softmax over sequence.  CDNA5 (gfx1250) wave32 WMMA implementation.
//
//   kernel 1: xprep   - fp32 [B,S,64] -> f16 time-major [S,B,64]  (BW bound)
//   kernel 2: rnn     - 32 blocks x 256 thr = 8 waves: 4 layer-stages x 2
//                       N-halves, software-pipelined over time (stage l works
//                       on timestep ep-l during epoch ep).  Weights pinned in
//                       VGPRs as WMMA B operands; one ping-pong LDS buffer per
//                       stage carries BOTH the inter-stage handoff and the
//                       recurrence; hardware v_tanh_f32; LDS-only barriers.
//   kernel 3: softmax - 512 blocks, softmax over S per batch row.
// ---------------------------------------------------------------------------

typedef __attribute__((ext_vector_type(16))) _Float16 v16h;
typedef __attribute__((ext_vector_type(8)))  float    v8f;

union V16 { v16h v; _Float16 h[16]; uint4 q[2]; };

constexpr int B  = 512;
constexpr int S  = 1024;
constexpr int IN = 64;
constexpr int H  = 32;
constexpr int L  = 4;
constexpr int MT = 16;   // batch rows per block (WMMA M-tile)

#define DEVFN __device__ __forceinline__

// ---- x: fp32 [B,S,IN] -> f16 time-major [S,B,IN] ---------------------------
__global__ void __launch_bounds__(256)
xprep_kernel(const float* __restrict__ x, _Float16* __restrict__ xh) {
  size_t idx = (size_t)blockIdx.x * 256 + threadIdx.x;
  if (idx >= (size_t)B * S * IN) return;
  int t = (int)(idx / (B * IN));
  int r = (int)(idx % (B * IN));
  int b = r / IN;
  int k = r % IN;
  xh[idx] = (_Float16)x[(size_t)b * S * IN + (size_t)t * IN + k];
}

// ---- hardware tanh (V_TANH_F32, TRANS pipe, co-executes with WMMA) ---------
DEVFN float fast_tanh(float x) {
#if __has_builtin(__builtin_amdgcn_tanhf)
  return __builtin_amdgcn_tanhf(x);                 // v_tanh_f32
#else
  float xc = fminf(fmaxf(x, -10.0f), 10.0f);
  float e  = __expf(2.0f * xc);
  return 1.0f - 2.0f * __builtin_amdgcn_rcpf(e + 1.0f);
#endif
}

// ---- LDS-only split barrier: wait own ds stores, then workgroup barrier ----
// (avoids __syncthreads' s_wait_storecnt on the global logit stores)
DEVFN void lds_barrier() {
  asm volatile("s_wait_dscnt 0x0\n\t"
               "s_barrier_signal -1\n\t"
               "s_barrier_wait -1" ::: "memory");
}

// ---- WMMA B operand (16-bit 32x16 K x N tile) built from W[n][k] (fp32) ----
// Lane n<16: col n, K in {0..7, 16..23}; lane>=16: col n-16, K in {8..15,24..31}
DEVFN V16 load_b_tile(const float* __restrict__ W, int ldk, int n_off, int k_off, int lane) {
  int n  = n_off + (lane & 15);
  int kb = (lane < 16) ? 0 : 8;
  V16 r;
#pragma unroll
  for (int i = 0; i < 8; ++i) r.h[i]     = (_Float16)W[n * ldk + k_off + kb + i];
#pragma unroll
  for (int i = 0; i < 8; ++i) r.h[8 + i] = (_Float16)W[n * ldk + k_off + 16 + kb + i];
  return r;
}

// ---- WMMA A operand (16-bit 16x32 M x K tile) from a plain [m][k] image ----
DEVFN V16 load_a_g(const _Float16* __restrict__ base, int ldk, int k_off, int lane) {
  int m  = lane & 15;
  int kb = (lane < 16) ? 0 : 8;
  V16 r;
  r.q[0] = *(const uint4*)(base + m * ldk + k_off + kb);
  r.q[1] = *(const uint4*)(base + m * ldk + k_off + 16 + kb);
  return r;
}

DEVFN V16 load_a_lds(const _Float16* base, int lane) {
  int m  = lane & 15;
  int kb = (lane < 16) ? 0 : 8;
  V16 r;
  r.q[0] = *(const uint4*)(base + m * H + kb);
  r.q[1] = *(const uint4*)(base + m * H + 16 + kb);
  return r;
}

#define WMMA(Av, Bv, Cv) \
  __builtin_amdgcn_wmma_f32_16x16x32_f16(false, (Av), false, (Bv), (short)0, (Cv), false, false)

// ---- pipelined recurrent kernel --------------------------------------------
// 8 waves: wave w -> stage (w>>1) = RNN layer, N-half (w&1).
// Epoch ep: stage l processes timestep t = ep - l (if 0 <= t < S).
// out_s[l][p] ping-pongs by epoch parity and carries both the recurrent state
// (stage l reads parity (ep+1)&1 = last epoch's write) and the inter-stage
// handoff (stage l+1 reads the same slot).  Writes go to parity ep&1.
// Phase structure per epoch:  [reads] barrier [compute+writes] barrier.
__global__ void __launch_bounds__(256)
rnn_kernel(const _Float16* __restrict__ xh,     // [S,B,IN] f16
           const float* __restrict__ h_init,    // [L,B,H]
           const float* __restrict__ Wih0,      // [H,IN]
           const float* __restrict__ WihR,      // [L-1,H,H]
           const float* __restrict__ Whh,       // [L,H,H]
           const float* __restrict__ bih,       // [L,H]
           const float* __restrict__ bhh,       // [L,H]
           const float* __restrict__ fcw,       // [1,H]
           const float* __restrict__ fcb,       // [1]
           float* __restrict__ logits,          // [B,S] scratch
           float* __restrict__ hfin)            // [L,B,H] (d_out tail)
{
  __shared__ __align__(16) _Float16 out_s[L][2][MT][H];  // per-stage ping-pong
  __shared__ float fcw_s[H];
  __shared__ float fcb_s;

  const int tid   = threadIdx.x;
  const int lane  = tid & 31;
  const int stage = __builtin_amdgcn_readfirstlane(tid >> 6);   // scalar: 0..3
  const int n_off = ((tid >> 5) & 1) * 16;                       // N half
  const int b0    = blockIdx.x * MT;

  // init both parity slots from h_state input (zeros in reference)
  for (int i = tid; i < L * MT * H; i += 256) {
    int l = i / (MT * H);
    int r = (i / H) % MT;
    int c = i % H;
    _Float16 hv = (_Float16)h_init[(size_t)l * B * H + (size_t)(b0 + r) * H + c];
    out_s[l][0][r][c] = hv;
    out_s[l][1][r][c] = hv;
  }
  if (tid < H) fcw_s[tid] = fcw[tid];
  if (tid == 0) fcb_s = fcb[0];

  // --- this stage's weight tiles, pinned in VGPRs as WMMA B operands ---
  V16 w0, w1, wr;
  if (stage == 0) {
    w0 = load_b_tile(Wih0, IN, n_off, 0,  lane);
    w1 = load_b_tile(Wih0, IN, n_off, 32, lane);
  } else {
    w0 = load_b_tile(WihR + (stage - 1) * H * H, H, n_off, 0, lane);
    w1 = w0;
  }
  wr = load_b_tile(Whh + stage * H * H, H, n_off, 0, lane);

  float bias;
  { int n = n_off + (lane & 15); bias = bih[stage * H + n] + bhh[stage * H + n]; }

  const int col   = n_off + (lane & 15);   // C-layout: lane -> output column
  const int rbase = (lane < 16) ? 0 : 8;   // C-layout: vgpr j -> row j/j+8

  __syncthreads();

  for (int ep = 0; ep < S + L; ++ep) {
    const int  t   = ep - stage;
    const bool act = __builtin_amdgcn_readfirstlane((int)((unsigned)t < (unsigned)S)) != 0;
    const int  rd  = (ep + 1) & 1;         // parity written last epoch

    // ================= read phase =================
    V16 a0, a1, ah;
    if (act) {
      if (stage == 0) {
        const _Float16* xrow = xh + (size_t)t * B * IN + (size_t)b0 * IN;
        a0 = load_a_g(xrow, IN, 0,  lane);
        a1 = load_a_g(xrow, IN, 32, lane);
        if (t + 1 < S)
          __builtin_prefetch(xh + (size_t)(t + 1) * B * IN + (size_t)(b0 + (lane & 15)) * IN, 0, 0);
      } else {
        a0 = load_a_lds(&out_s[stage - 1][rd][0][0], lane);
      }
      ah = load_a_lds(&out_s[stage][rd][0][0], lane);
    }

    // fused per-timestep Linear(32->1): h3 for timestep ep-4 was committed
    // at the end of epoch ep-1; read it here (barrier-protected).
    {
      const int tl = ep - L;
      if (tid >= 192 && tid < 192 + MT && tl >= 0 && tl < S) {
        int row = tid - 192;
        float s = fcb_s;
#pragma unroll
        for (int k = 0; k < H; ++k) s += fcw_s[k] * (float)out_s[3][rd][row][k];
        logits[(size_t)(b0 + row) * S + tl] = s;
      }
    }

    lds_barrier();   // reads done; writers may proceed

    // ================= compute + write phase =================
    if (act) {
      v8f c;
#pragma unroll
      for (int j = 0; j < 8; ++j) c[j] = bias;
      c = WMMA(a0.v, w0.v, c);
      if (stage == 0) c = WMMA(a1.v, w1.v, c);
      c = WMMA(ah.v, wr.v, c);

      _Float16* dst = &out_s[stage][ep & 1][0][0];
#pragma unroll
      for (int j = 0; j < 8; ++j) {
        float v = fast_tanh(c[j]);
        dst[(j + rbase) * H + col] = (_Float16)v;   // unconditional, branch-free
      }
    }

    lds_barrier();   // writes visible for next epoch's readers
  }

  // h_finals: stage l's last write (t = S-1 at epoch S-1+l) sits in parity
  // (S-1+l)&1 = (l+1)&1 (S-1 odd).  Cooperative copy, f16 -> f32.
  for (int i = tid; i < L * MT * H; i += 256) {
    int l = i / (MT * H);
    int r = (i / H) % MT;
    int c = i % H;
    hfin[(size_t)l * B * H + (size_t)(b0 + r) * H + c] =
        (float)out_s[l][(l + 1) & 1][r][c];
  }
}

// ---- softmax over S per batch row ------------------------------------------
__global__ void __launch_bounds__(256)
softmax_kernel(const float* __restrict__ logits, float* __restrict__ probs) {
  __shared__ float red[256];
  const int b = blockIdx.x;
  const int tid = threadIdx.x;
  const float* lrow = logits + (size_t)b * S;

  float l0[4];
  float m = -3.402823466e38f;
#pragma unroll
  for (int i = 0; i < 4; ++i) { l0[i] = lrow[tid + i * 256]; m = fmaxf(m, l0[i]); }
  red[tid] = m; __syncthreads();
  for (int off = 128; off > 0; off >>= 1) {
    if (tid < off) red[tid] = fmaxf(red[tid], red[tid + off]);
    __syncthreads();
  }
  m = red[0]; __syncthreads();

  float e[4], s = 0.f;
#pragma unroll
  for (int i = 0; i < 4; ++i) { e[i] = __expf(l0[i] - m); s += e[i]; }
  red[tid] = s; __syncthreads();
  for (int off = 128; off > 0; off >>= 1) {
    if (tid < off) red[tid] += red[tid + off];
    __syncthreads();
  }
  float inv = __builtin_amdgcn_rcpf(red[0]);   // v_rcp_f32, no IEEE div expansion
#pragma unroll
  for (int i = 0; i < 4; ++i) probs[(size_t)b * S + tid + i * 256] = e[i] * inv;
}

// ---------------------------------------------------------------------------
extern "C" void kernel_launch(void* const* d_in, const int* in_sizes, int n_in,
                              void* d_out, int out_size, void* d_ws, size_t ws_size,
                              hipStream_t stream) {
  (void)in_sizes; (void)n_in; (void)out_size; (void)ws_size;

  const float* x    = (const float*)d_in[0];
  const float* h0   = (const float*)d_in[1];
  const float* Wih0 = (const float*)d_in[2];
  const float* WihR = (const float*)d_in[3];
  const float* Whh  = (const float*)d_in[4];
  const float* bih  = (const float*)d_in[5];
  const float* bhh  = (const float*)d_in[6];
  const float* fcw  = (const float*)d_in[7];
  const float* fcb  = (const float*)d_in[8];

  float* out = (float*)d_out;                 // [B*S] probs, then [L*B*H] h_finals
  char*  ws  = (char*)d_ws;

  // workspace: f16 x image (64 MiB) + fp32 logits (2 MiB)
  const size_t xh_bytes = (size_t)B * S * IN * 2;
  _Float16* xh     = (_Float16*)ws;
  float*    logits = (float*)(ws + ((xh_bytes + 255) & ~(size_t)255));

  const size_t total = (size_t)B * S * IN;
  xprep_kernel<<<(unsigned)((total + 255) / 256), 256, 0, stream>>>(x, xh);
  rnn_kernel<<<B / MT, 256, 0, stream>>>(xh, h0, Wih0, WihR, Whh, bih, bhh,
                                         fcw, fcb, logits, out + (size_t)B * S);
  softmax_kernel<<<B, 256, 0, stream>>>(logits, out);
}
